// MultiHeadAttention_21328807592315
// MI455X (gfx1250) — matrix operand order repaired and checked by reading, not compiled, not run on hardware
//
#include <hip/hip_runtime.h>
#include <hip/hip_bf16.h>

// MHA forward for MI455X (gfx1250, wave32, WMMA + TDM).
// Pipeline: [gemm+rope -> q,k (bf16)] [gemm -> v (bf16)] [flash attention] [gemm -> out (f32)]

typedef __attribute__((ext_vector_type(16))) __bf16 v16bf;
typedef __attribute__((ext_vector_type(8)))  float  v8f;
typedef __attribute__((ext_vector_type(4)))  unsigned int u32x4;
typedef __attribute__((ext_vector_type(8)))  int i32x8;
typedef __attribute__((ext_vector_type(4)))  int i32x4;

#define D_MODEL 1024
#define SEQ     2048
#define BATCH   4
#define HEADS   16
#define DH      64
#define NROWS   (BATCH * SEQ)   // 8192

__device__ __forceinline__ unsigned short f2bf(float f) {
    unsigned int u = __float_as_uint(f);
    u += 0x7FFFu + ((u >> 16) & 1u);     // round to nearest even
    return (unsigned short)(u >> 16);
}

union FragB { uint4 u[2]; v16bf v; };

__device__ __forceinline__ v8f wmma_bf16(const v16bf& a, const v16bf& b, v8f c) {
    return __builtin_amdgcn_wmma_f32_16x16x32_bf16(false, a, false, b, (short)0, c, false, false);
}

__device__ __forceinline__ unsigned lds_off_u32(const void* p) {
    // generic pointer to LDS: aperture lives in the high 32 bits, offset in the low 32
    return (unsigned)(unsigned long long)(size_t)p;
}

// TDM 2-D tile load: global (bf16, row stride strideElems) -> LDS (with optional row padding).
// padIntervalCode: pad after 2<<code DWORDs; padAmountCode: pad by (code+1) DWORDs.
__device__ __forceinline__ void tdm_load_2d_bf16(unsigned ldsOff, const void* gptr,
                                                 unsigned tileD0, unsigned tileD1,
                                                 unsigned long long strideElems,
                                                 unsigned padIntervalCode,
                                                 unsigned padAmountCode,
                                                 unsigned padEnable)
{
    unsigned long long ga = (unsigned long long)(size_t)gptr;
    u32x4 g0;
    g0[0] = 1u;                                            // count=1, user descriptor
    g0[1] = ldsOff;                                        // lds_addr
    g0[2] = (unsigned)(ga & 0xFFFFFFFFull);                // global_addr[31:0]
    g0[3] = (unsigned)((ga >> 32) & 0x01FFFFFFull) | (2u << 30);  // global_addr[56:32], type=2
    i32x8 g1;
    g1[0] = (int)((1u << 16) | (padEnable << 20) |
                  (padIntervalCode << 22) | (padAmountCode << 25)); // data_size=2B, pads
    g1[1] = (int)((tileD0 & 0xFFFFu) << 16);               // tensor_dim0[15:0] (== tileD0)
    g1[2] = (int)(((tileD0 >> 16) & 0xFFFFu) | ((tileD1 & 0xFFFFu) << 16)); // tensor_dim1 lo
    g1[3] = (int)(((tileD1 >> 16) & 0xFFFFu) | ((tileD0 & 0xFFFFu) << 16)); // tile_dim0
    g1[4] = (int)(tileD1 & 0xFFFFu);                       // tile_dim1, tile_dim2=0
    g1[5] = (int)(strideElems & 0xFFFFFFFFull);            // tensor_dim0_stride[31:0]
    g1[6] = (int)((strideElems >> 32) & 0xFFFFull);        // stride[47:32], dim1_stride=0
    g1[7] = 0;
    i32x4 z4 = {0, 0, 0, 0};
#if defined(__clang_major__) && (__clang_major__ >= 23)
    i32x8 z8 = {0, 0, 0, 0, 0, 0, 0, 0};
    __builtin_amdgcn_tensor_load_to_lds(g0, g1, z4, z4, z8, 0);
#else
    __builtin_amdgcn_tensor_load_to_lds(g0, g1, z4, z4, 0);
#endif
}

// ---------------------------------------------------------------------------
// Tiled GEMM: Out[m,e] = sum_d A[m,d] * W[e,d]   (A: NROWSxD, W: DxD row-major)
// Workgroup tile 128(M) x 64(N), 8 waves, each wave 32x32 (2x2 WMMA tiles).
// ---------------------------------------------------------------------------
template<bool A_BF16, bool ROPE, bool OUT_F32>
__global__ __launch_bounds__(256)
void gemm_xw_kernel(const void* __restrict__ Aptr,
                    const float* __restrict__ W,
                    void* __restrict__ Outptr,
                    const int* __restrict__ positions)
{
    __shared__ unsigned short lA[128 * 40];   // 32 cols + 8 pad (keeps 16B align)
    __shared__ unsigned short lB[64 * 40];

    const int tid  = threadIdx.x;
    const int lane = tid & 31;
    const int wid  = tid >> 5;
    const int half = lane >> 4;
    const int lm   = lane & 15;
    const int wm   = wid & 3;                 // wave M slot (0..3) -> 32 rows
    const int wn   = wid >> 2;                // wave N slot (0..1) -> 32 cols
    const int tileN = blockIdx.x * 64;
    const int tileM = blockIdx.y * 128;

    v8f acc[2][2] = {};

    for (int k0 = 0; k0 < D_MODEL; k0 += 32) {
        // ---- stage A tile (128 x 32) as bf16 ----
        if (A_BF16) {
            // A already bf16 in workspace: one TDM descriptor moves the whole tile,
            // padding each 16-DWORD row (32 bf16) by 4 DWORDs -> LDS stride 40 elems.
            if (wid == 0) {
                const unsigned short* Ab = (const unsigned short*)Aptr;
                tdm_load_2d_bf16(lds_off_u32(lA),
                                 Ab + (size_t)tileM * D_MODEL + k0,
                                 /*tile*/32, 128, /*stride*/D_MODEL,
                                 /*interval 16 DW*/3, /*pad 4 DW*/3, 1);
            }
        } else {
            const float* Af = (const float*)Aptr;
#pragma unroll
            for (int i = 0; i < 4; ++i) {
                int c   = tid + 256 * i;              // 1024 chunks of 4 floats
                int row = c >> 3;
                int col = (c & 7) * 4;
                float4 d = *(const float4*)(Af + (size_t)(tileM + row) * D_MODEL + k0 + col);
                uint2 p;
                p.x = (unsigned)f2bf(d.x) | ((unsigned)f2bf(d.y) << 16);
                p.y = (unsigned)f2bf(d.z) | ((unsigned)f2bf(d.w) << 16);
                *(uint2*)(&lA[row * 40 + col]) = p;
            }
        }
        // ---- stage B tile (64 x 32) from W (always f32) ----
#pragma unroll
        for (int i = 0; i < 2; ++i) {
            int c   = tid + 256 * i;                  // 512 chunks of 4 floats
            int row = c >> 3;
            int col = (c & 7) * 4;
            float4 d = *(const float4*)(W + (size_t)(tileN + row) * D_MODEL + k0 + col);
            uint2 p;
            p.x = (unsigned)f2bf(d.x) | ((unsigned)f2bf(d.y) << 16);
            p.y = (unsigned)f2bf(d.z) | ((unsigned)f2bf(d.w) << 16);
            *(uint2*)(&lB[row * 40 + col]) = p;
        }
        // ---- prefetch next k-step tiles ----
        if (k0 + 32 < D_MODEL) {
            if (tid < 128) {
                if (!A_BF16)
                    __builtin_prefetch((const float*)Aptr +
                                       (size_t)(tileM + tid) * D_MODEL + k0 + 32, 0, 0);
            } else if (tid < 192) {
                __builtin_prefetch(W + (size_t)(tileN + (tid - 128)) * D_MODEL + k0 + 32, 0, 0);
            }
        }
        if (A_BF16 && wid == 0) __builtin_amdgcn_s_wait_tensorcnt(0);
        __syncthreads();

        // ---- build fragments (contiguous k = 16*half + s, same in A and B) ----
        FragB a[2], b[2];
#pragma unroll
        for (int i = 0; i < 2; ++i) {
            const unsigned short* p = &lA[(wm * 32 + i * 16 + lm) * 40 + 16 * half];
            a[i].u[0] = *(const uint4*)(p);           // K = 16*half .. +7
            a[i].u[1] = *(const uint4*)(p + 8);       // K = 16*half+8 .. +15
        }
#pragma unroll
        for (int j = 0; j < 2; ++j) {
            const unsigned short* p = &lB[(wn * 32 + j * 16 + lm) * 40 + 16 * half];
            b[j].u[0] = *(const uint4*)(p);           // K = 16*half .. +7
            b[j].u[1] = *(const uint4*)(p + 8);       // K = 16*half+8 .. +15
        }
#pragma unroll
        for (int i = 0; i < 2; ++i)
#pragma unroll
            for (int j = 0; j < 2; ++j)
                acc[i][j] = wmma_bf16(a[i].v, b[j].v, acc[i][j]);
        __syncthreads();
    }

    // ---- epilogue: optional RoPE (pairs are adjacent lanes), store ----
#pragma unroll
    for (int i = 0; i < 2; ++i) {
#pragma unroll
        for (int jj = 0; jj < 8; ++jj) {
            int mg = tileM + wm * 32 + i * 16 + jj + 8 * half;
#pragma unroll
            for (int j = 0; j < 2; ++j) {
                int eg = tileN + wn * 32 + j * 16 + lm;
                float val = acc[i][j][jj];
                if (ROPE) {
                    float pos = (float)positions[mg & (SEQ - 1)];
                    int   fi  = (eg & (DH - 1)) >> 1;
                    float inv = __expf((float)fi * -0.28782313662425572f); // ln(1e4)/32
                    float ang = pos * inv;
                    float sn, cs;
                    __sincosf(ang, &sn, &cs);
                    float other = __shfl_xor(val, 1);
                    val = (eg & 1) ? fmaf(other, sn, val * cs)
                                   : fmaf(val, cs, -(other * sn));
                }
                if (OUT_F32)
                    ((float*)Outptr)[(size_t)mg * D_MODEL + eg] = val;
                else
                    ((unsigned short*)Outptr)[(size_t)mg * D_MODEL + eg] = f2bf(val);
            }
        }
    }
}

// ---------------------------------------------------------------------------
// Flash attention: grid (S/64, H, B), 128 threads = 4 waves, each wave 16 rows.
// ---------------------------------------------------------------------------
__global__ __launch_bounds__(128)
void attn_kernel(const unsigned short* __restrict__ Q,
                 const unsigned short* __restrict__ K,
                 const unsigned short* __restrict__ V,
                 unsigned short* __restrict__ O)
{
    __shared__ unsigned short lK[32 * 72];        // keys x dh, row-major (TDM-filled)
    __shared__ unsigned short lVT[64 * 40];       // dh x keys (transposed)
    __shared__ unsigned short lP[4][16 * 40];     // per-wave P tile

    const int tid  = threadIdx.x;
    const int lane = tid & 31;
    const int wave = tid >> 5;
    const int half = lane >> 4;
    const int lm   = lane & 15;

    const int qb0 = blockIdx.x * 64;
    const int h   = blockIdx.y;
    const int b   = blockIdx.z;
    const int qr0 = qb0 + wave * 16;
    const size_t rowBase = (size_t)b * SEQ;

    // Q fragments held in registers for the whole kernel
    FragB aq[2];
    {
        const unsigned short* qrow = Q + (rowBase + qr0 + lm) * D_MODEL + h * DH;
#pragma unroll
        for (int dk = 0; dk < 2; ++dk) {
            const unsigned short* p = qrow + dk * 32 + 16 * half;
            aq[dk].u[0] = *(const uint4*)(p);
            aq[dk].u[1] = *(const uint4*)(p + 8);
        }
    }

    v8f oacc[4] = {};
    float m_i[8], l_i[8];
#pragma unroll
    for (int jj = 0; jj < 8; ++jj) { m_i[jj] = -1e30f; l_i[jj] = 0.f; }

    const int nkb = (qb0 + 64) / 32;
    for (int kb = 0; kb < nkb; ++kb) {
        const int kb0 = kb * 32;

        // ---- TDM: K tile (32 rows x 64 bf16, row pad 32 DW -> +4 DW = stride 72) ----
        if (wave == 0) {
            tdm_load_2d_bf16(lds_off_u32(lK),
                             K + (rowBase + kb0) * D_MODEL + h * DH,
                             /*tile*/DH, 32, /*stride*/D_MODEL,
                             /*interval 32 DW*/4, /*pad 4 DW*/3, 1);
        }

        // ---- stage V transposed (64 x 32) manually ----
#pragma unroll
        for (int i = 0; i < 4; ++i) {
            int c  = tid + 128 * i;                    // 512 chunks of 4 bf16
            int r  = c >> 4;
            int dc = (c & 15) * 4;
            uint2 vv = *(const uint2*)(V + (rowBase + kb0 + r) * D_MODEL + h * DH + dc);
            lVT[(dc + 0) * 40 + r] = (unsigned short)(vv.x & 0xFFFFu);
            lVT[(dc + 1) * 40 + r] = (unsigned short)(vv.x >> 16);
            lVT[(dc + 2) * 40 + r] = (unsigned short)(vv.y & 0xFFFFu);
            lVT[(dc + 3) * 40 + r] = (unsigned short)(vv.y >> 16);
        }

        // ---- prefetch next block's K and V rows ----
        if (kb + 1 < nkb) {
            int r = tid & 31;
            const unsigned short* nb = ((tid & 32) ? V : K) +
                                       (rowBase + kb0 + 32 + r) * D_MODEL + h * DH;
            if (tid < 64) __builtin_prefetch(nb, 0, 0);
        }

        if (wave == 0) __builtin_amdgcn_s_wait_tensorcnt(0);
        __syncthreads();

        // ---- scores S = Q K^T (two 16x16 tiles, K-dim 64 = 2 chained WMMAs) ----
        v8f s0 = {}, s1 = {};
#pragma unroll
        for (int dk = 0; dk < 2; ++dk) {
            FragB bk0, bk1;
            const unsigned short* p0 = &lK[(lm) * 72 + dk * 32 + 16 * half];
            bk0.u[0] = *(const uint4*)p0;  bk0.u[1] = *(const uint4*)(p0 + 8);
            const unsigned short* p1 = &lK[(16 + lm) * 72 + dk * 32 + 16 * half];
            bk1.u[0] = *(const uint4*)p1;  bk1.u[1] = *(const uint4*)(p1 + 8);
            s0 = wmma_bf16(aq[dk].v, bk0.v, s0);
            s1 = wmma_bf16(aq[dk].v, bk1.v, s1);
        }

        // ---- causal mask + online softmax (16-lane butterfly reductions) ----
        float p0v[8], p1v[8], alpha[8];
#pragma unroll
        for (int jj = 0; jj < 8; ++jj) {
            int   qg  = qr0 + jj + 8 * half;
            float sv0 = s0[jj] * 0.125f;
            float sv1 = s1[jj] * 0.125f;
            if (kb0 + lm > qg)      sv0 = -1e30f;
            if (kb0 + 16 + lm > qg) sv1 = -1e30f;
            float mx = fmaxf(sv0, sv1);
            mx = fmaxf(mx, __shfl_xor(mx, 1));
            mx = fmaxf(mx, __shfl_xor(mx, 2));
            mx = fmaxf(mx, __shfl_xor(mx, 4));
            mx = fmaxf(mx, __shfl_xor(mx, 8));
            float mnew = fmaxf(m_i[jj], mx);
            alpha[jj]  = __expf(m_i[jj] - mnew);
            m_i[jj]    = mnew;
            p0v[jj] = __expf(sv0 - mnew);
            p1v[jj] = __expf(sv1 - mnew);
            float rs = p0v[jj] + p1v[jj];
            rs += __shfl_xor(rs, 1);
            rs += __shfl_xor(rs, 2);
            rs += __shfl_xor(rs, 4);
            rs += __shfl_xor(rs, 8);
            l_i[jj] = l_i[jj] * alpha[jj] + rs;
        }
#pragma unroll
        for (int jj = 0; jj < 8; ++jj) {
            oacc[0][jj] *= alpha[jj]; oacc[1][jj] *= alpha[jj];
            oacc[2][jj] *= alpha[jj]; oacc[3][jj] *= alpha[jj];
        }

        // ---- P: C-layout -> A-layout via per-wave LDS tile ----
        unsigned short* pw = &lP[wave][0];
#pragma unroll
        for (int jj = 0; jj < 8; ++jj) {
            int r = jj + 8 * half;
            pw[r * 40 + lm]      = f2bf(p0v[jj]);
            pw[r * 40 + 16 + lm] = f2bf(p1v[jj]);
        }
        FragB ap;
        {
            const unsigned short* p = pw + lm * 40 + 16 * half;
            ap.u[0] = *(const uint4*)p;
            ap.u[1] = *(const uint4*)(p + 8);
        }

        // ---- O += P V (4 output d-tiles, one WMMA each) ----
#pragma unroll
        for (int t4 = 0; t4 < 4; ++t4) {
            FragB bv;
            const unsigned short* p = &lVT[(t4 * 16 + lm) * 40 + 16 * half];
            bv.u[0] = *(const uint4*)p;
            bv.u[1] = *(const uint4*)(p + 8);
            oacc[t4] = wmma_bf16(ap.v, bv.v, oacc[t4]);
        }
        __syncthreads();
    }

    // ---- finalize: divide by row sums, store bf16 ----
#pragma unroll
    for (int jj = 0; jj < 8; ++jj) {
        float inv = 1.0f / l_i[jj];
        int   mg  = qr0 + jj + 8 * half;
        size_t base = (rowBase + mg) * D_MODEL + h * DH;
#pragma unroll
        for (int t4 = 0; t4 < 4; ++t4)
            O[base + t4 * 16 + lm] = f2bf(oacc[t4][jj] * inv);
    }
}

// ---------------------------------------------------------------------------
extern "C" void kernel_launch(void* const* d_in, const int* in_sizes, int n_in,
                              void* d_out, int out_size, void* d_ws, size_t ws_size,
                              hipStream_t stream)
{
    (void)in_sizes; (void)n_in; (void)out_size; (void)ws_size;
    const float* x   = (const float*)d_in[0];
    const float* w_q = (const float*)d_in[1];
    const float* w_k = (const float*)d_in[2];
    const float* w_v = (const float*)d_in[3];
    const float* w_o = (const float*)d_in[4];
    const int*   pos = (const int*)d_in[5];

    const size_t mat = (size_t)NROWS * D_MODEL;      // 8M bf16 elements
    unsigned short* q_b = (unsigned short*)d_ws;
    unsigned short* k_b = q_b + mat;
    unsigned short* v_b = k_b + mat;
    unsigned short* a_b = v_b + mat;                 // total 64 MB of workspace

    dim3 gG(D_MODEL / 64, NROWS / 128);              // (16, 64)
    gemm_xw_kernel<false, true,  false><<<gG, 256, 0, stream>>>(x,   w_q, q_b,  pos);
    gemm_xw_kernel<false, true,  false><<<gG, 256, 0, stream>>>(x,   w_k, k_b,  pos);
    gemm_xw_kernel<false, false, false><<<gG, 256, 0, stream>>>(x,   w_v, v_b,  pos);

    attn_kernel<<<dim3(SEQ / 64, HEADS, BATCH), 128, 0, stream>>>(q_b, k_b, v_b, a_b);

    gemm_xw_kernel<true,  false, true ><<<gG, 256, 0, stream>>>(a_b, w_o, d_out, pos);
}
